// HEGN_45990509805767
// MI455X (gfx1250) — compile-verified, matrix operand order checked by reading, not run on hardware
//
#include <hip/hip_runtime.h>
#include <stdint.h>

#define BB 4
#define NN 2048
#define CC 32
#define KNN 16
#define D96 96
#define M3N (3*NN)
#define EPSF 1e-6f
#define BNEPS 1e-5f
#define NSLOPE 0.2f

typedef float v8f __attribute__((ext_vector_type(8)));
typedef __bf16 v16bf __attribute__((ext_vector_type(16)));
typedef unsigned int u32x4 __attribute__((ext_vector_type(4)));
typedef int i32x4 __attribute__((ext_vector_type(4)));
typedef int i32x8 __attribute__((ext_vector_type(8)));

union BF16V { v16bf v; unsigned short u[16]; };

#define IDX4(b,c,d,n) (((((size_t)(b))*CC+(c))*3+(d))*NN+(n))

__device__ inline unsigned short f2bf(float f) {
  unsigned u = __float_as_uint(f);
  return (unsigned short)((u + 0x7FFFu + ((u >> 16) & 1u)) >> 16);
}

// CDNA5 16-bit A/B fragment K-pattern (ISA 7.12.2): lane group h (lane>=16),
// element e of v16bf -> K index within a 32-wide K step.
__device__ inline int kpat(int e, int h) {
  int v = e >> 1;
  int base = (v < 4) ? (2 * v) : (16 + 2 * (v - 4));
  return base + 8 * h + (e & 1);
}

__device__ inline void topk_insert(float v, int id, float* best, int* bidx) {
  if (v > best[KNN - 1]) {
    best[KNN - 1] = v; bidx[KNN - 1] = id;
#pragma unroll
    for (int j = KNN - 1; j > 0; --j) {
      if (best[j] > best[j - 1]) {
        float tv = best[j]; best[j] = best[j - 1]; best[j - 1] = tv;
        int ti = bidx[j]; bidx[j] = bidx[j - 1]; bidx[j - 1] = ti;
      } else break;
    }
  }
}

// vn_lrelu core for one channel's 3-vector (BN stats precomputed: mean, istd).
__device__ inline void vn_mix(const float* p0, const float* d3, float mean, float istd, float* out) {
  float nr = sqrtf(p0[0]*p0[0] + p0[1]*p0[1] + p0[2]*p0[2]) + EPSF;
  float s = ((nr - mean) * istd) / nr;
  float p[3] = { p0[0]*s, p0[1]*s, p0[2]*s };
  float dot = p[0]*d3[0] + p[1]*d3[1] + p[2]*d3[2];
  float dsq = d3[0]*d3[0] + d3[1]*d3[1] + d3[2]*d3[2];
  float coef = dot / (dsq + EPSF);
#pragma unroll
  for (int d = 0; d < 3; ++d) {
    float neg = p[d] - coef * d3[d];
    float selv = (dot >= 0.f) ? p[d] : neg;
    out[d] = NSLOPE * p[d] + (1.f - NSLOPE) * selv;
  }
}

// One-wave WMMA edge GEMM: P(32x48) = W(32x64) @ E(64x48), E/out in LDS.
// 2 M-tiles x 3 N-tiles x 2 K-steps = 12 v_wmma per call. Caller syncs.
__device__ inline void wmma_edge_gemm(const float* __restrict__ W,
                                      const float (*E)[48], float (*out)[48], int t) {
  const int h = t >> 4, r = t & 15;
#pragma unroll
  for (int m0 = 0; m0 < CC; m0 += 16) {
    BF16V a0, a1;
#pragma unroll
    for (int e2 = 0; e2 < 16; ++e2) {
      int k0 = kpat(e2, h);
      a0.u[e2] = f2bf(W[(m0 + r) * 64 + k0]);
      a1.u[e2] = f2bf(W[(m0 + r) * 64 + k0 + 32]);
    }
#pragma unroll
    for (int c0 = 0; c0 < 48; c0 += 16) {
      BF16V b0, b1;
#pragma unroll
      for (int e2 = 0; e2 < 16; ++e2) {
        int k0 = kpat(e2, h);
        b0.u[e2] = f2bf(E[k0][c0 + r]);
        b1.u[e2] = f2bf(E[k0 + 32][c0 + r]);
      }
      v8f acc = {};
      acc = __builtin_amdgcn_wmma_f32_16x16x32_bf16(false, a0.v, false, b0.v, (short)0, acc, false, false);
      acc = __builtin_amdgcn_wmma_f32_16x16x32_bf16(false, a1.v, false, b1.v, (short)0, acc, false, false);
#pragma unroll
      for (int i = 0; i < 8; ++i)
        out[m0 + i + 8 * h][c0 + r] = acc[i];
    }
  }
}

// ---------------- kNN over raw 3-D points (scalar; d=3 is too skinny for WMMA) ---------
__global__ void knn3_kernel(const float* __restrict__ P, int* __restrict__ idx3) {
  int gid = blockIdx.x * blockDim.x + threadIdx.x;
  if (gid >= BB * NN) return;
  int b = gid / NN, n = gid - b * NN;
  float p0 = P[((size_t)b*3 + 0)*NN + n];
  float p1 = P[((size_t)b*3 + 1)*NN + n];
  float p2 = P[((size_t)b*3 + 2)*NN + n];
  float best[KNN]; int bidx[KNN];
#pragma unroll
  for (int j = 0; j < KNN; ++j) { best[j] = -3.4e38f; bidx[j] = 0; }
  for (int m = 0; m < NN; ++m) {
    float d0 = P[((size_t)b*3 + 0)*NN + m] - p0;
    float d1 = P[((size_t)b*3 + 1)*NN + m] - p1;
    float d2 = P[((size_t)b*3 + 2)*NN + m] - p2;
    topk_insert(-(d0*d0 + d1*d1 + d2*d2), m, best, bidx);
  }
  for (int j = 0; j < KNN; ++j) idx3[(size_t)gid * KNN + j] = bidx[j];
}

// ---------------- DGCNN edge layer (Kin=2, scalar math; two-pass BN) -------------------
__global__ void dg_stats(const float* __restrict__ P, const int* __restrict__ idx3,
                         const float* __restrict__ Wf, float* __restrict__ st) {
  int bn = blockIdx.x; int b = bn / NN, n = bn - b * NN; int c = threadIdx.x;
  float xi[3];
#pragma unroll
  for (int d = 0; d < 3; ++d) xi[d] = P[((size_t)b*3 + d)*NN + n];
  float w0 = Wf[c*2 + 0], w1 = Wf[c*2 + 1];
  float s = 0.f, s2 = 0.f;
  for (int k = 0; k < KNN; ++k) {
    int j = idx3[(size_t)bn * KNN + k];
    float nr2 = 0.f;
#pragma unroll
    for (int d = 0; d < 3; ++d) {
      float xj = P[((size_t)b*3 + d)*NN + j];
      float p = w0 * (xj - xi[d]) + w1 * xi[d];
      nr2 += p * p;
    }
    float nr = sqrtf(nr2) + EPSF;
    s += nr; s2 += nr * nr;
  }
  atomicAdd(&st[2*c], s);
  atomicAdd(&st[2*c + 1], s2);
}

__global__ void bn_finalize(float* st, float cnt) {
  int c = threadIdx.x;
  float mean = st[2*c] / cnt;
  float var  = st[2*c + 1] / cnt - mean * mean;
  st[2*c] = mean;
  st[2*c + 1] = rsqrtf(var + BNEPS);
}

__global__ void dg_apply(const float* __restrict__ P, const int* __restrict__ idx3,
                         const float* __restrict__ Wf, const float* __restrict__ Wd,
                         const float* __restrict__ st, float* __restrict__ out) {
  int bn = blockIdx.x; int b = bn / NN, n = bn - b * NN; int c = threadIdx.x;
  float xi[3];
#pragma unroll
  for (int d = 0; d < 3; ++d) xi[d] = P[((size_t)b*3 + d)*NN + n];
  float wf0 = Wf[c*2 + 0], wf1 = Wf[c*2 + 1];
  float wd0 = Wd[c*2 + 0], wd1 = Wd[c*2 + 1];
  float mean = st[2*c], istd = st[2*c + 1];
  float acc[3] = {0.f, 0.f, 0.f};
  for (int k = 0; k < KNN; ++k) {
    int j = idx3[(size_t)bn * KNN + k];
    float p0[3], d3[3], r[3];
#pragma unroll
    for (int d = 0; d < 3; ++d) {
      float xj = P[((size_t)b*3 + d)*NN + j];
      float e0 = xj - xi[d], e1 = xi[d];
      p0[d] = wf0 * e0 + wf1 * e1;
      d3[d] = wd0 * e0 + wd1 * e1;
    }
    vn_mix(p0, d3, mean, istd, r);
#pragma unroll
    for (int d = 0; d < 3; ++d) acc[d] += r[d];
  }
#pragma unroll
  for (int d = 0; d < 3; ++d) out[IDX4(b, c, d, n)] = acc[d] * (1.f / KNN);
}

// ---------------- generic VN channel GEMM via WMMA bf16 (f32 accumulate) ---------------
// Out[b, 32, 6144] = W(32 x Kin) @ In[b, Kin, 6144]
__global__ void vn_gemm_wmma(const float* __restrict__ W, const float* __restrict__ In,
                             float* __restrict__ Out, int Kin) {
  __shared__ float lds_w[16 * 64];
  __shared__ float lds_in[64 * 16];
  const int t = threadIdx.x;
  const int col0 = blockIdx.x * 16;
  const int m0 = blockIdx.y * 16;
  const int b = blockIdx.z;
  for (int i = t; i < 16 * Kin; i += 32)
    lds_w[i] = W[(m0 + i / Kin) * Kin + (i % Kin)];
  for (int i = t; i < Kin * 16; i += 32) {
    int kk = i >> 4, c2 = i & 15;
    lds_in[i] = In[((size_t)b * Kin + kk) * M3N + col0 + c2];
  }
  __syncthreads();
  const int h = t >> 4, r = t & 15;
  v8f acc = {};
  for (int s = 0; s * 32 < Kin; ++s) {
    BF16V a, bm;
#pragma unroll
    for (int e = 0; e < 16; ++e) {
      int k = kpat(e, h) + 32 * s;
      a.u[e]  = f2bf(lds_w[r * Kin + k]);
      bm.u[e] = f2bf(lds_in[k * 16 + r]);
    }
    acc = __builtin_amdgcn_wmma_f32_16x16x32_bf16(false, a.v, false, bm.v, (short)0, acc, false, false);
  }
#pragma unroll
  for (int i = 0; i < 8; ++i) {
    int m = m0 + i + 8 * h;
    Out[((size_t)b * CC + m) * M3N + col0 + r] = acc[i];
  }
}

// ---------------- 96-dim squared norms (for the -||x||^2 distance terms) ----------------
__global__ void sqnorm96(const float* __restrict__ F, float* __restrict__ xx) {
  int gid = blockIdx.x * blockDim.x + threadIdx.x;
  if (gid >= BB * NN) return;
  int b = gid / NN, n = gid - b * NN;
  float s = 0.f;
  for (int k = 0; k < D96; ++k) {
    float v = F[((size_t)b * D96 + k) * NN + n];
    s += v * v;
  }
  xx[gid] = s;
}

// ---------------- fused 96-dim kNN: WMMA Gram tiles + TDM B-tile loads + top-16 --------
__global__ void knn96_wmma(const float* __restrict__ F, const float* __restrict__ xx,
                           int* __restrict__ idxk) {
  __shared__ float lds_a[16 * D96];
  __shared__ float lds_b[D96 * 16];
  __shared__ float sc[16 * 16];
  const int t = threadIdx.x;
  const int m0 = blockIdx.x * 16;
  const int b = blockIdx.y;
  const float* Fb = F + (size_t)b * D96 * NN;
  for (int i = t; i < 16 * D96; i += 32) {
    int r = i / D96, k = i - r * D96;
    lds_a[i] = Fb[(size_t)k * NN + m0 + r];
  }
  __syncthreads();
  const int h = t >> 4, r = t & 15;
  BF16V afr[3];
#pragma unroll
  for (int s = 0; s < 3; ++s)
#pragma unroll
    for (int e = 0; e < 16; ++e)
      afr[s].u[e] = f2bf(lds_a[r * D96 + kpat(e, h) + 32 * s]);

  float best[KNN]; int bidx[KNN];
#pragma unroll
  for (int j = 0; j < KNN; ++j) { best[j] = -3.4e38f; bidx[j] = 0; }

  unsigned lds_off = (unsigned)(size_t)(&lds_b[0]);  // flat->LDS: low 32 bits are the LDS offset
  for (int n0 = 0; n0 < NN; n0 += 16) {
    // --- Tensor Data Mover: DMA a 96x16 f32 tile (row stride NN) into LDS ---
    unsigned long long ga = (unsigned long long)(const void*)(Fb + n0);
    u32x4 g0;
    g0[0] = 1u;                                   // count=1 valid descriptor
    g0[1] = lds_off;                              // lds_addr
    g0[2] = (unsigned)ga;                         // global_addr lo
    g0[3] = (unsigned)((ga >> 32) & 0x1FFFFFFull) | (2u << 30);  // addr hi | type=2
    i32x8 g1;
    g1[0] = (int)(2u << 16);                      // data_size = 4 bytes
    g1[1] = (int)((NN & 0xFFFF) << 16);           // tensor_dim0 lo16
    g1[2] = (int)(((NN >> 16) & 0xFFFF) | ((D96 & 0xFFFF) << 16)); // dim0 hi | dim1 lo
    g1[3] = (int)(16 << 16);                      // dim1 hi=0 | tile_dim0=16
    g1[4] = (int)(D96);                           // tile_dim1=96 | tile_dim2=0
    g1[5] = (int)(NN);                            // tensor_dim0_stride lo32
    g1[6] = 0;
    g1[7] = 0;
    i32x4 gz = {0, 0, 0, 0};
#if defined(__clang_major__) && (__clang_major__ >= 23)
    i32x8 gz8 = {0, 0, 0, 0, 0, 0, 0, 0};
    __builtin_amdgcn_tensor_load_to_lds(g0, g1, gz, gz, gz8, 0);
#else
    __builtin_amdgcn_tensor_load_to_lds(g0, g1, gz, gz, 0);
#endif
    __builtin_amdgcn_s_wait_tensorcnt(0);
    __syncthreads();
    __builtin_prefetch(Fb + n0 + 16, 0, 0);       // global_prefetch_b8 next tile

    v8f acc = {};
#pragma unroll
    for (int s = 0; s < 3; ++s) {
      BF16V bm;
#pragma unroll
      for (int e = 0; e < 16; ++e) {
        int k = kpat(e, h) + 32 * s;
        bm.u[e] = f2bf(lds_b[k * 16 + r]);
      }
      acc = __builtin_amdgcn_wmma_f32_16x16x32_bf16(false, afr[s].v, false, bm.v, (short)0, acc, false, false);
    }
#pragma unroll
    for (int i = 0; i < 8; ++i) {
      int row = i + 8 * h;
      sc[row * 16 + r] = 2.f * acc[i] - xx[b * NN + m0 + row] - xx[b * NN + n0 + r];
    }
    __syncthreads();
    if (t < 16) {
      for (int c2 = 0; c2 < 16; ++c2)
        topk_insert(sc[t * 16 + c2], n0 + c2, best, bidx);
    }
    __syncthreads();
  }
  if (t < 16)
    for (int j = 0; j < KNN; ++j)
      idxk[((size_t)b * NN + m0 + t) * KNN + j] = bidx[j];
}

// ---------------- BN stats over (B,32,3,N): per-channel sum/sumsq of ||.||+eps ---------
__global__ void bn_stats3(const float* __restrict__ p, float* __restrict__ st) {
  int c = blockIdx.x, t = threadIdx.x;
  __shared__ float sh[256], sh2[256];
  float s = 0.f, s2 = 0.f;
  for (int i = t + blockIdx.y * 256; i < BB * NN; i += 256 * gridDim.y) {
    int b = i / NN, n = i - b * NN;
    float a0 = p[IDX4(b, c, 0, n)];
    float a1 = p[IDX4(b, c, 1, n)];
    float a2 = p[IDX4(b, c, 2, n)];
    float nr = sqrtf(a0*a0 + a1*a1 + a2*a2) + EPSF;
    s += nr; s2 += nr * nr;
  }
  sh[t] = s; sh2[t] = s2; __syncthreads();
  for (int o = 128; o > 0; o >>= 1) {
    if (t < o) { sh[t] += sh[t + o]; sh2[t] += sh2[t + o]; }
    __syncthreads();
  }
  if (t == 0) { atomicAdd(&st[2*c], sh[0]); atomicAdd(&st[2*c + 1], sh2[0]); }
}

// ---------------- apply BN + lrelu (+ optional chnorm) on (B,32,3,N) ------------------
__global__ void vn_combine3(const float* __restrict__ p, const float* __restrict__ dbuf,
                            const float* __restrict__ st, float* __restrict__ out,
                            int do_chnorm) {
  int bn = blockIdx.x; int b = bn / NN, n = bn - b * NN; int c = threadIdx.x;
  __shared__ float sh[CC];
  float p0[3], d3[3], r[3];
#pragma unroll
  for (int d = 0; d < 3; ++d) { p0[d] = p[IDX4(b,c,d,n)]; d3[d] = dbuf[IDX4(b,c,d,n)]; }
  vn_mix(p0, d3, st[2*c], st[2*c + 1], r);
  if (do_chnorm) {
    float nr2 = r[0]*r[0] + r[1]*r[1] + r[2]*r[2];
    sh[c] = nr2; __syncthreads();
    float nn2 = 0.f;
    for (int i = 0; i < CC; ++i) nn2 += sh[i];
    float nnv = sqrtf(nn2), nr = sqrtf(nr2);
    float scale = (nr / fmaxf(nnv, 1e-12f)) / fmaxf(nr, 1e-12f);
#pragma unroll
    for (int d = 0; d < 3; ++d) r[d] *= scale;
  }
#pragma unroll
  for (int d = 0; d < 3; ++d) out[IDX4(b, c, d, n)] = r[d];
}

// ---------------- BN stats for the K/V edge paths: WMMA edge GEMM ----------------------
__global__ void kv_stats(const float* __restrict__ Y, const int* __restrict__ idxk,
                         const float* __restrict__ Wkf, const float* __restrict__ Wvf,
                         float* __restrict__ stK, float* __restrict__ stV) {
  int bn = blockIdx.x; int b = bn / NN, n = bn - b * NN; int c = threadIdx.x;
  __shared__ float e[64][48];
  __shared__ float pk[CC][48];
  __shared__ float pv[CC][48];
  float yc[3];
#pragma unroll
  for (int d = 0; d < 3; ++d) yc[d] = Y[IDX4(b, c, d, n)];
  for (int k = 0; k < KNN; ++k) {
    int j = idxk[(size_t)bn * KNN + k];
#pragma unroll
    for (int d = 0; d < 3; ++d) {
      float yj = Y[IDX4(b, c, d, j)];
      e[c][k*3 + d] = yj - yc[d];
      e[CC + c][k*3 + d] = yc[d];
    }
  }
  __syncthreads();
  wmma_edge_gemm(Wkf, e, pk, c);
  wmma_edge_gemm(Wvf, e, pv, c);
  __syncthreads();
  float sk = 0.f, sk2 = 0.f, sv = 0.f, sv2 = 0.f;
  for (int k = 0; k < KNN; ++k) {
    float nk = sqrtf(pk[c][3*k]*pk[c][3*k] + pk[c][3*k+1]*pk[c][3*k+1] + pk[c][3*k+2]*pk[c][3*k+2]) + EPSF;
    float nv = sqrtf(pv[c][3*k]*pv[c][3*k] + pv[c][3*k+1]*pv[c][3*k+1] + pv[c][3*k+2]*pv[c][3*k+2]) + EPSF;
    sk += nk; sk2 += nk*nk; sv += nv; sv2 += nv*nv;
  }
  atomicAdd(&stK[2*c], sk);  atomicAdd(&stK[2*c+1], sk2);
  atomicAdd(&stV[2*c], sv);  atomicAdd(&stV[2*c+1], sv2);
}

// ---------------- cross-context attention core: WMMA edge GEMMs -----------------------
__global__ void attn_kernel(const float* __restrict__ X, const float* __restrict__ Y,
                            const int* __restrict__ idxk, const float* __restrict__ QX,
                            const float* __restrict__ Wkf, const float* __restrict__ Wkd,
                            const float* __restrict__ Wvf, const float* __restrict__ Wvd,
                            const float* __restrict__ stK, const float* __restrict__ stV,
                            float* __restrict__ out) {
  int bn = blockIdx.x; int b = bn / NN, n = bn - b * NN; int c = threadIdx.x;
  __shared__ float e[64][48];
  __shared__ float pk[CC][48];
  __shared__ float dk[CC][48];
  __shared__ float qk[CC][KNN];
  __shared__ float attw[4][KNN];
  __shared__ float red[CC];
  float yc[3], q3[3];
#pragma unroll
  for (int d = 0; d < 3; ++d) { yc[d] = Y[IDX4(b,c,d,n)]; q3[d] = QX[IDX4(b,c,d,n)]; }
  float mk = stK[2*c], ik = stK[2*c+1], mv = stV[2*c], iv = stV[2*c+1];

  // build the 64x48 edge matrix once (16 neighbors x 3 dims)
  for (int k = 0; k < KNN; ++k) {
    int j = idxk[(size_t)bn * KNN + k];
#pragma unroll
    for (int d = 0; d < 3; ++d) {
      float yj = Y[IDX4(b, c, d, j)];
      e[c][k*3 + d] = yj - yc[d];
      e[CC + c][k*3 + d] = yc[d];
    }
  }
  __syncthreads();

  // pass A: Ky = BN+lrelu+chnorm( Wk_f@e , Wk_d@e ), then qk = <Ky, Qx>
  wmma_edge_gemm(Wkf, e, pk, c);
  wmma_edge_gemm(Wkd, e, dk, c);
  __syncthreads();
  for (int k = 0; k < KNN; ++k) {
    float r[3];
    vn_mix(&pk[c][3*k], &dk[c][3*k], mk, ik, r);
    float nr2 = r[0]*r[0] + r[1]*r[1] + r[2]*r[2];
    red[c] = nr2; __syncthreads();
    float nn2 = 0.f;
    for (int i = 0; i < CC; ++i) nn2 += red[i];
    float nnv = sqrtf(nn2), nr = sqrtf(nr2);
    float scale = (nr / fmaxf(nnv, 1e-12f)) / fmaxf(nr, 1e-12f);
    float dot = 0.f;
#pragma unroll
    for (int d = 0; d < 3; ++d) dot += (r[d] * scale) * q3[d];
    qk[c][k] = dot;
    __syncthreads();
  }

  // per-head softmax over K
  if (c < 4) {
    const float invs = 0.20412414523193154f;  // 1/sqrt(3*HEAD_C)
    float lg[KNN], mx = -3.4e38f;
    for (int k = 0; k < KNN; ++k) {
      float s = 0.f;
      for (int u = 0; u < 8; ++u) s += qk[c*8 + u][k];
      lg[k] = s * invs; mx = fmaxf(mx, lg[k]);
    }
    float se = 0.f;
    for (int k = 0; k < KNN; ++k) { lg[k] = expf(lg[k] - mx); se += lg[k]; }
    for (int k = 0; k < KNN; ++k) attw[c][k] = lg[k] / se;
  }
  __syncthreads();

  // pass B: Vy = BN+lrelu( Wv_f@e , Wv_d@e ), weighted sum with attention
  wmma_edge_gemm(Wvf, e, pk, c);
  wmma_edge_gemm(Wvd, e, dk, c);
  __syncthreads();
  float acc[3] = {0.f, 0.f, 0.f};
  for (int k = 0; k < KNN; ++k) {
    float r[3];
    vn_mix(&pk[c][3*k], &dk[c][3*k], mv, iv, r);
    float a = attw[c >> 3][k];
#pragma unroll
    for (int d = 0; d < 3; ++d) acc[d] += a * r[d];
  }
#pragma unroll
  for (int d = 0; d < 3; ++d)
    out[IDX4(b, c, d, n)] = X[IDX4(b, c, d, n)] + acc[d];
}

// ---------------- global stage helpers -------------------------------------------------
__global__ void mean_over_n(const float* __restrict__ f, float* __restrict__ out) {
  int gid = blockIdx.x * blockDim.x + threadIdx.x;
  if (gid >= BB * CC * 3) return;
  size_t base = (size_t)gid * NN;
  float s = 0.f;
  for (int n = 0; n < NN; ++n) s += f[base + n];
  out[gid] = s / (float)NN;
}

__global__ void concat64(const float* __restrict__ f, const float* __restrict__ fm,
                         float* __restrict__ out64) {
  size_t gid = (size_t)blockIdx.x * blockDim.x + threadIdx.x;
  if (gid >= (size_t)BB * 64 * 3 * NN) return;
  size_t n = gid % NN; size_t rest = gid / NN;
  int d = (int)(rest % 3); rest /= 3;
  int c = (int)(rest % 64); int b = (int)(rest / 64);
  out64[gid] = (c < CC) ? f[IDX4(b, c, d, n)] : fm[((size_t)b * CC + (c - CC)) * 3 + d];
}

__global__ void par_kernel(const float* __restrict__ g, float* __restrict__ par) {
  int gid = blockIdx.x * blockDim.x + threadIdx.x;
  if (gid >= BB * NN) return;
  int b = gid / NN, n = gid - b * NN;
  float m[3];
#pragma unroll
  for (int d = 0; d < 3; ++d) {
    float s = 0.f;
    for (int c = 0; c < CC; ++c) s += g[IDX4(b, c, d, n)];
    m[d] = s / (float)CC;
  }
  float nr = sqrtf(m[0]*m[0] + m[1]*m[1] + m[2]*m[2]);
#pragma unroll
  for (int d = 0; d < 3; ++d) par[((size_t)b * 3 + d) * NN + n] = m[d] / nr;
}

__global__ void logit_kernel(const float* __restrict__ gx, const float* __restrict__ gy,
                             const float* __restrict__ px, const float* __restrict__ py,
                             float* __restrict__ lg) {
  int gid = blockIdx.x * blockDim.x + threadIdx.x;
  if (gid >= BB * NN) return;
  int b = gid / NN, n = gid - b * NN;
  float acc = 0.f;
  for (int c = 0; c < CC; ++c) {
    float ax = 0.f, ay = 0.f;
#pragma unroll
    for (int d = 0; d < 3; ++d) {
      ax += gx[IDX4(b, c, d, n)] * px[((size_t)b*3 + d)*NN + n];
      ay += gy[IDX4(b, c, d, n)] * py[((size_t)b*3 + d)*NN + n];
    }
    acc += ax * ay;
  }
  lg[gid] = acc;
}

__global__ void rank_select(const float* __restrict__ lg, int* __restrict__ sel) {
  int gid = blockIdx.x * blockDim.x + threadIdx.x;
  if (gid >= BB * NN) return;
  int b = gid / NN, n = gid - b * NN;
  float v = lg[(size_t)b * NN + n];
  int rk = 0;
  for (int m = 0; m < NN; ++m) {
    float u = lg[(size_t)b * NN + m];
    rk += (u > v) || (u == v && m < n);
  }
  if (rk < NN / 4) sel[b * (NN / 4) + rk] = n;
}

__global__ void gather_mean(const float* __restrict__ g, const int* __restrict__ sel,
                            float* __restrict__ out) {
  int gid = blockIdx.x * blockDim.x + threadIdx.x;
  if (gid >= BB * CC * 3) return;
  int b = gid / (CC * 3); int cd = gid - b * (CC * 3);
  int c = cd / 3, d = cd - c * 3;
  const int NS4 = NN / 4;
  float s = 0.f;
  for (int j = 0; j < NS4; ++j) {
    int n = sel[b * NS4 + j];
    s += g[IDX4(b, c, d, n)];
  }
  out[gid] = s / (float)NS4;
}

// ---------------- final: Wh VN-lrelu (batch-only BN), 3x3 SVD, R & S ------------------
__global__ void final_kernel(const float* __restrict__ Whf, const float* __restrict__ Whd,
                             const float* __restrict__ fxs, const float* __restrict__ fys,
                             float* __restrict__ dout) {
  __shared__ float nx[BB][CC], ny[BB][CC];
  __shared__ float Fx[BB][CC][3], Fy[BB][CC][3];
  int t = threadIdx.x; int b = t >> 5; int c = t & 31;
  float px[3] = {0,0,0}, dx3[3] = {0,0,0}, py[3] = {0,0,0}, dy3[3] = {0,0,0};
  for (int i = 0; i < CC; ++i) {
    float wf = Whf[c*CC + i], wd = Whd[c*CC + i];
#pragma unroll
    for (int d = 0; d < 3; ++d) {
      float vx = fxs[((size_t)b*CC + i)*3 + d];
      float vy = fys[((size_t)b*CC + i)*3 + d];
      px[d] += wf * vx; dx3[d] += wd * vx;
      py[d] += wf * vy; dy3[d] += wd * vy;
    }
  }
  nx[b][c] = sqrtf(px[0]*px[0]+px[1]*px[1]+px[2]*px[2]) + EPSF;
  ny[b][c] = sqrtf(py[0]*py[0]+py[1]*py[1]+py[2]*py[2]) + EPSF;
  __syncthreads();
  float mx = 0.f, m2x = 0.f, my = 0.f, m2y = 0.f;
  for (int bb2 = 0; bb2 < BB; ++bb2) {
    float v = nx[bb2][c]; mx += v; m2x += v * v;
    v = ny[bb2][c]; my += v; m2y += v * v;
  }
  mx /= BB; my /= BB;
  float ix = rsqrtf(m2x / BB - mx * mx + BNEPS);
  float iy = rsqrtf(m2y / BB - my * my + BNEPS);
  float rx[3], ry[3];
  vn_mix(px, dx3, mx, ix, rx);
  vn_mix(py, dy3, my, iy, ry);
#pragma unroll
  for (int d = 0; d < 3; ++d) { Fx[b][c][d] = rx[d]; Fy[b][c][d] = ry[d]; }
  __syncthreads();

  if (t < BB) {
    int bb2 = t;
    float H[3][3];
    for (int d = 0; d < 3; ++d)
      for (int e2 = 0; e2 < 3; ++e2) {
        float s = 0.f;
        for (int cc2 = 0; cc2 < CC; ++cc2) s += Fx[bb2][cc2][d] * Fy[bb2][cc2][e2];
        H[d][e2] = s;
      }
    // Jacobi on A = H^T H
    float A[3][3], V[3][3] = {{1,0,0},{0,1,0},{0,0,1}};
    for (int d = 0; d < 3; ++d)
      for (int e2 = 0; e2 < 3; ++e2) {
        float s = 0.f;
        for (int f2 = 0; f2 < 3; ++f2) s += H[f2][d] * H[f2][e2];
        A[d][e2] = s;
      }
    const int PP[3] = {0, 0, 1}, QQ[3] = {1, 2, 2};
    for (int sweep = 0; sweep < 12; ++sweep) {
      for (int pi = 0; pi < 3; ++pi) {
        int p = PP[pi], q = QQ[pi];
        float apq = A[p][q];
        if (fabsf(apq) > 1e-12f) {
          float tau = (A[q][q] - A[p][p]) / (2.f * apq);
          float tt = ((tau >= 0.f) ? 1.f : -1.f) / (fabsf(tau) + sqrtf(1.f + tau * tau));
          float cs = rsqrtf(1.f + tt * tt), sn = tt * cs;
          for (int i = 0; i < 3; ++i) {
            float aip = A[i][p], aiq = A[i][q];
            A[i][p] = cs * aip - sn * aiq; A[i][q] = sn * aip + cs * aiq;
          }
          for (int i = 0; i < 3; ++i) {
            float api = A[p][i], aqi = A[q][i];
            A[p][i] = cs * api - sn * aqi; A[q][i] = sn * api + cs * aqi;
          }
          for (int i = 0; i < 3; ++i) {
            float vip = V[i][p], viq = V[i][q];
            V[i][p] = cs * vip - sn * viq; V[i][q] = sn * vip + cs * viq;
          }
        }
      }
    }
    float lam[3] = {A[0][0], A[1][1], A[2][2]};
    for (int i = 0; i < 2; ++i)
      for (int j = i + 1; j < 3; ++j)
        if (lam[j] > lam[i]) {
          float tl = lam[i]; lam[i] = lam[j]; lam[j] = tl;
          for (int r2 = 0; r2 < 3; ++r2) {
            float tv = V[r2][i]; V[r2][i] = V[r2][j]; V[r2][j] = tv;
          }
        }
    float sv[3], U[3][3];
    for (int k = 0; k < 3; ++k) {
      sv[k] = sqrtf(fmaxf(lam[k], 0.f));
      for (int i = 0; i < 3; ++i) {
        float hv = 0.f;
        for (int j = 0; j < 3; ++j) hv += H[i][j] * V[j][k];
        U[i][k] = hv / fmaxf(sv[k], 1e-12f);
      }
    }
    for (int i = 0; i < 3; ++i)
      for (int j = 0; j < 3; ++j) {
        float s = 0.f;
        for (int k = 0; k < 3; ++k) s += U[i][k] * V[k][j];
        dout[bb2 * 9 + i * 3 + j] = s;
      }
    for (int d = 0; d < 3; ++d) {
      float sx = 0.f, sy = 0.f;
      for (int cc2 = 0; cc2 < CC; ++cc2) {
        sx += Fx[bb2][cc2][d] * Fx[bb2][cc2][d];
        sy += Fy[bb2][cc2][d] * Fy[bb2][cc2][d];
      }
      dout[BB * 9 + bb2 * 3 + d] = sqrtf(sy) / sqrtf(sx);
    }
  }
}

// ---------------- host-side pipeline ---------------------------------------------------
static void run_cc(const float* Xsrc, const float* Ysrc, float* Out,
                   const float* Wq_f, const float* Wq_d, const float* Wk_f, const float* Wk_d,
                   const float* Wv_f, const float* Wv_d,
                   float* pbuf, float* dbuf, float* qx, float* xxb, int* idxk,
                   float* stq, float* stk, float* stv, hipStream_t s) {
  dim3 gg(M3N / 16, 2, BB);
  vn_gemm_wmma<<<gg, 32, 0, s>>>(Wq_f, Xsrc, pbuf, CC);
  vn_gemm_wmma<<<gg, 32, 0, s>>>(Wq_d, Xsrc, dbuf, CC);
  bn_stats3<<<dim3(CC, 4), 256, 0, s>>>(pbuf, stq);
  bn_finalize<<<1, CC, 0, s>>>(stq, (float)(BB * NN));
  vn_combine3<<<BB * NN, CC, 0, s>>>(pbuf, dbuf, stq, qx, 1);
  sqnorm96<<<(BB * NN + 255) / 256, 256, 0, s>>>(Ysrc, xxb);
  knn96_wmma<<<dim3(NN / 16, BB), 32, 0, s>>>(Ysrc, xxb, idxk);
  kv_stats<<<BB * NN, CC, 0, s>>>(Ysrc, idxk, Wk_f, Wv_f, stk, stv);
  bn_finalize<<<1, CC, 0, s>>>(stk, (float)(BB * NN * KNN));
  bn_finalize<<<1, CC, 0, s>>>(stv, (float)(BB * NN * KNN));
  attn_kernel<<<BB * NN, CC, 0, s>>>(Xsrc, Ysrc, idxk, qx, Wk_f, Wk_d, Wv_f, Wv_d, stk, stv, Out);
}

static void run_g(const float* Fin, const float* Wg_f, const float* Wg_d,
                  float* fmean, float* in64, float* pbuf, float* dbuf,
                  float* stg, float* gout, hipStream_t s) {
  mean_over_n<<<(BB * CC * 3 + 63) / 64, 64, 0, s>>>(Fin, fmean);
  concat64<<<(int)(((size_t)BB * 64 * 3 * NN + 255) / 256), 256, 0, s>>>(Fin, fmean, in64);
  dim3 gg(M3N / 16, 2, BB);
  vn_gemm_wmma<<<gg, 32, 0, s>>>(Wg_f, in64, pbuf, 64);
  vn_gemm_wmma<<<gg, 32, 0, s>>>(Wg_d, in64, dbuf, 64);
  bn_stats3<<<dim3(CC, 4), 256, 0, s>>>(pbuf, stg);
  bn_finalize<<<1, CC, 0, s>>>(stg, (float)(BB * NN));
  vn_combine3<<<BB * NN, CC, 0, s>>>(pbuf, dbuf, stg, gout, 0);
}

extern "C" void kernel_launch(void* const* d_in, const int* in_sizes, int n_in,
                              void* d_out, int out_size, void* d_ws, size_t ws_size,
                              hipStream_t stream) {
  (void)in_sizes; (void)n_in; (void)out_size; (void)ws_size;
  const float* x    = (const float*)d_in[0];
  const float* y    = (const float*)d_in[1];
  const float* Wdg_f= (const float*)d_in[2];
  const float* Wdg_d= (const float*)d_in[3];
  const float* Wq_f = (const float*)d_in[4];
  const float* Wq_d = (const float*)d_in[5];
  const float* Wk_f = (const float*)d_in[6];
  const float* Wk_d = (const float*)d_in[7];
  const float* Wv_f = (const float*)d_in[8];
  const float* Wv_d = (const float*)d_in[9];
  const float* Wg_f = (const float*)d_in[10];
  const float* Wg_d = (const float*)d_in[11];
  const float* Wh_f = (const float*)d_in[12];
  const float* Wh_d = (const float*)d_in[13];

  float* ws = (float*)d_ws;
  const size_t S = (size_t)BB * CC * 3 * NN;
  float* fx   = ws;
  float* fy   = ws + 1 * S;
  float* fx2  = ws + 2 * S;
  float* fy2  = ws + 3 * S;
  float* qx   = ws + 4 * S;
  float* pbuf = ws + 5 * S;
  float* dbuf = ws + 6 * S;
  float* in64 = ws + 7 * S;          // 2*S
  float* gx   = ws + 9 * S;
  float* gy   = ws + 10 * S;
  float* misc = ws + 11 * S;
  float* xxb   = misc;                           // BB*NN
  float* parx  = xxb + (size_t)BB * NN;          // BB*3*NN
  float* pary  = parx + (size_t)BB * 3 * NN;
  float* logit = pary + (size_t)BB * 3 * NN;     // BB*NN
  float* fmean = logit + (size_t)BB * NN;        // BB*CC*3
  float* fxs   = fmean + (size_t)BB * CC * 3;
  float* fys   = fxs + (size_t)BB * CC * 3;
  float* stats = fys + (size_t)BB * CC * 3;      // 16 slots x 64 floats
  int* ib    = (int*)(stats + 16 * 64);
  int* idx3x = ib;
  int* idx3y = idx3x + (size_t)BB * NN * KNN;
  int* idxk  = idx3y + (size_t)BB * NN * KNN;
  int* sel   = idxk + (size_t)BB * NN * KNN;

  hipMemsetAsync(stats, 0, 16 * 64 * sizeof(float), stream);

  // ---- layer 1: DGCNN edge conv on raw points -----------------------------------------
  knn3_kernel<<<(BB * NN + 255) / 256, 256, 0, stream>>>(x, idx3x);
  knn3_kernel<<<(BB * NN + 255) / 256, 256, 0, stream>>>(y, idx3y);
  dg_stats<<<BB * NN, CC, 0, stream>>>(x, idx3x, Wdg_f, stats + 0 * 64);
  bn_finalize<<<1, CC, 0, stream>>>(stats + 0 * 64, (float)(BB * NN * KNN));
  dg_apply<<<BB * NN, CC, 0, stream>>>(x, idx3x, Wdg_f, Wdg_d, stats + 0 * 64, fx);
  dg_stats<<<BB * NN, CC, 0, stream>>>(y, idx3y, Wdg_f, stats + 1 * 64);
  bn_finalize<<<1, CC, 0, stream>>>(stats + 1 * 64, (float)(BB * NN * KNN));
  dg_apply<<<BB * NN, CC, 0, stream>>>(y, idx3y, Wdg_f, Wdg_d, stats + 1 * 64, fy);

  // ---- cross contexts -----------------------------------------------------------------
  run_cc(fx, fy, fx2, Wq_f, Wq_d, Wk_f, Wk_d, Wv_f, Wv_d,
         pbuf, dbuf, qx, xxb, idxk, stats + 2 * 64, stats + 3 * 64, stats + 4 * 64, stream);
  run_cc(fy, fx2, fy2, Wq_f, Wq_d, Wk_f, Wk_d, Wv_f, Wv_d,
         pbuf, dbuf, qx, xxb, idxk, stats + 5 * 64, stats + 6 * 64, stats + 7 * 64, stream);

  // ---- global mixing (Wg) -------------------------------------------------------------
  run_g(fx2, Wg_f, Wg_d, fmean, in64, pbuf, dbuf, stats + 8 * 64, gx, stream);
  run_g(fy2, Wg_f, Wg_d, fmean, in64, pbuf, dbuf, stats + 9 * 64, gy, stream);

  // ---- invariant scoring, top-N/4 selection, pooled heads -----------------------------
  par_kernel<<<(BB * NN + 255) / 256, 256, 0, stream>>>(gx, parx);
  par_kernel<<<(BB * NN + 255) / 256, 256, 0, stream>>>(gy, pary);
  logit_kernel<<<(BB * NN + 255) / 256, 256, 0, stream>>>(gx, gy, parx, pary, logit);
  rank_select<<<(BB * NN + 255) / 256, 256, 0, stream>>>(logit, sel);
  gather_mean<<<(BB * CC * 3 + 63) / 64, 64, 0, stream>>>(gx, sel, fxs);
  gather_mean<<<(BB * CC * 3 + 63) / 64, 64, 0, stream>>>(gy, sel, fys);

  // ---- Wh head + 3x3 SVD --> (R, S) ---------------------------------------------------
  final_kernel<<<1, BB * CC, 0, stream>>>(Wh_f, Wh_d, fxs, fys, (float*)d_out);
}
